// VariationalLSTM_61194694033738
// MI455X (gfx1250) — matrix operand order repaired and checked by reading, not compiled, not run on hardware
//
#include <hip/hip_runtime.h>
#include <hip/hip_bf16.h>
#include <stdint.h>

// Problem constants
#define TT 1024
#define BB 128
#define HH 256
#define II 256
#define NG 1024     // 4*H gate columns

typedef __bf16 bf16x16 __attribute__((ext_vector_type(16)));
typedef float  f32x8   __attribute__((ext_vector_type(8)));
typedef unsigned int u32x8v __attribute__((ext_vector_type(8)));
typedef unsigned int u32x4  __attribute__((ext_vector_type(4)));
typedef int i32x8 __attribute__((ext_vector_type(8)));
typedef int i32x4 __attribute__((ext_vector_type(4)));

#if defined(__HIP_DEVICE_COMPILE__) && __has_builtin(__builtin_amdgcn_tensor_load_to_lds) && __has_builtin(__builtin_amdgcn_s_wait_tensorcnt)
#define HAVE_TDM 1
#else
#define HAVE_TDM 0
#endif

__device__ inline float sigm(float x) { return 1.0f / (1.0f + __expf(-x)); }
__device__ inline float tanh_f(float x) {
    x = fminf(fmaxf(x, -15.0f), 15.0f);
    float e = __expf(-2.0f * x);
    return (1.0f - e) / (1.0f + e);
}

__device__ inline bf16x16 bc16(uint4 a, uint4 b) {
    u32x8v u = {a.x, a.y, a.z, a.w, b.x, b.y, b.z, b.w};
    return __builtin_bit_cast(bf16x16, u);
}

__device__ inline f32x8 splat8(float s) {
    f32x8 r;
#pragma unroll
    for (int i = 0; i < 8; ++i) r[i] = s;
    return r;
}

#if HAVE_TDM
// 2-D TDM tile load: 16 rows x 256 bf16, global row stride = TT*II elements,
// packed contiguously into LDS. One instruction per step per workgroup.
// 6-arg builtin form: (g0, g1, g2, g3, g4, cpol); groups 2..4 unused for 2-D.
__device__ inline void tdm_load_tile(const __bf16* gsrc, unsigned lds_off) {
    unsigned long long ga = (unsigned long long)(uintptr_t)gsrc;
    u32x4 g0;
    g0.x = 1u;                                            // count=1, user mode
    g0.y = lds_off;                                       // LDS byte address
    g0.z = (unsigned)ga;                                  // global_addr lo32
    g0.w = (unsigned)((ga >> 32) & 0x1FFFFFFu) | (2u << 30); // hi bits + type=2
    i32x8 g1;
    g1[0] = (int)(1u << 16);        // data_size = 1 (2 bytes/elem)
    g1[1] = (int)(256u << 16);      // tensor_dim0 = 256 (bits 79:48 lo half)
    g1[2] = (int)(16u << 16);       // tensor_dim1 = 16
    g1[3] = (int)(256u << 16);      // tile_dim0 = 256
    g1[4] = 16;                     // tile_dim1 = 16, tile_dim2 = 0
    g1[5] = (int)(TT * II);         // tensor_dim0_stride lo32 = 262144 elems
    g1[6] = 0;
    g1[7] = 0;
    i32x4 z4 = {0, 0, 0, 0};        // 2-D tensor: groups 2/3 unused
    i32x8 z8 = {0, 0, 0, 0, 0, 0, 0, 0};
    __builtin_amdgcn_tensor_load_to_lds(g0, g1, z4, z4, z8, 0);
}
#endif

// One K=32 slice: A fragment from LDS, 4 gate-tile B fragments from L2-hot
// packed weights, 4 WMMAs. Kept inside a rolled loop so weight loads cannot be
// hoisted across the time loop (that caused scratch spills when fully unrolled).
__device__ __forceinline__ void gemm_k32(const __bf16* __restrict__ abase, int kl,
                                         const __bf16* __restrict__ wk,
                                         int r, int hi, int lane, int jn,
                                         f32x8 acc[4]) {
    const uint4* ap = (const uint4*)(abase + r * 256 + kl + hi * 8);
    uint4 a0 = ap[0];
    uint4 a1 = ap[2];  // +16 elements
    bf16x16 Af = bc16(a0, a1);
    const __bf16* wcol = wk + jn + r;  // lane holds column jn+r, K rows in table order
#pragma unroll
    for (int gi = 0; gi < 4; ++gi) {
        bf16x16 Bf;
#pragma unroll
        for (int s = 0; s < 16; ++s)
            Bf[s] = wcol[(size_t)((s & 7) + hi * 8 + (s >> 3) * 16) * NG + gi * HH];
        acc[gi] = __builtin_amdgcn_wmma_f32_16x16x32_bf16(
            false, Af, false, Bf, (short)0, acc[gi], false, false);
    }
}

// ---- prep kernel 1: fp32 x -> bf16, same [B,T,I] layout -------------------
__global__ void cvt_x_kernel(const float* __restrict__ x, __bf16* __restrict__ xb) {
    size_t i = ((size_t)blockIdx.x * blockDim.x + threadIdx.x) * 4;
    float4 v = *(const float4*)(x + i);
    xb[i + 0] = (__bf16)v.x;
    xb[i + 1] = (__bf16)v.y;
    xb[i + 2] = (__bf16)v.z;
    xb[i + 3] = (__bf16)v.w;
}

// ---- prep kernel 2: WcatT[l][k][g] = (k<256 ? W_ih[l][g][k] : W_hh[l][g][k-256])
// K-major so a B-fragment K row is contiguous across lanes. Also fuse bias = b_ih+b_hh.
__global__ void pack_w_kernel(const float* __restrict__ Wih, const float* __restrict__ Whh,
                              const float* __restrict__ bih, const float* __restrict__ bhh,
                              __bf16* __restrict__ WT, float* __restrict__ bias) {
    int idx = blockIdx.x * blockDim.x + threadIdx.x;  // over 2*512*1024
    int g = idx & 1023;
    int k = (idx >> 10) & 511;
    int l = idx >> 19;
    float v = (k < II) ? Wih[((size_t)(l * NG + g)) * II + k]
                       : Whh[((size_t)(l * NG + g)) * HH + (k - II)];
    WT[idx] = (__bf16)v;
    if (k == 0) bias[l * NG + g] = bih[l * NG + g] + bhh[l * NG + g];
}

// ---- persistent LSTM layer kernel -----------------------------------------
// grid = 8 blocks (16-batch slice each), block = 512 threads = 16 waves.
// Wave w computes gate columns {w,w+16,w+32,w+48}*16 == i/f/g/o for h cols [16w,16w+16)
// so the full cell update runs in accumulator layout with c carried in VGPRs.
__global__ __launch_bounds__(512) void lstm_layer_kernel(
    const __bf16* __restrict__ xin,   // [B,T,256] bf16 input sequence
    __bf16* __restrict__ ybf,         // bf16 masked output (layer 0) or null
    float* __restrict__ yf32,         // fp32 masked output (layer 1) or null
    const __bf16* __restrict__ WT,    // [512][1024] bf16, K-major packed weights
    const float* __restrict__ bias,   // [1024] fused bias
    const float* __restrict__ masks_l)// [3][B][H] (out, h, c) for this layer
{
    __shared__ __align__(16) __bf16 sX[2][16 * 256];  // double-buffered x_t tile
    __shared__ __align__(16) __bf16 sH[2][16 * 256];  // double-buffered carried h

    const int tid = threadIdx.x;
    const int lane = tid & 31;
    const int wv = tid >> 5;        // wave 0..15
    const int hi = lane >> 4;       // half-wave select
    const int r = lane & 15;        // A row / C column-in-tile
    const int b0 = blockIdx.x * 16; // batch slice base
    const int jn = wv * 16;         // h-column tile base for this wave
    const int jw = jn + r;          // this lane's h column

    // zero carried-h buffers
    {
        __bf16* hp = &sH[0][0];
        for (int i = tid; i < 2 * 16 * 256; i += 512) hp[i] = (__bf16)0.0f;
    }

    // per-lane constants: masks (accumulator layout) and fused bias per gate
    float mo[8], mh[8], mc[8], c_reg[8];
#pragma unroll
    for (int v = 0; v < 8; ++v) {
        int b = b0 + v + hi * 8;
        mo[v] = masks_l[(0 * BB + b) * HH + jw];
        mh[v] = masks_l[(1 * BB + b) * HH + jw];
        mc[v] = masks_l[(2 * BB + b) * HH + jw];
        c_reg[v] = 0.0f;
    }
    float bg[4];
#pragma unroll
    for (int gi = 0; gi < 4; ++gi) bg[gi] = bias[gi * HH + jw];

    __syncthreads();

#if HAVE_TDM
    if (wv == 0)
        tdm_load_tile(xin + (size_t)(b0 * TT + 0) * II, (unsigned)(uintptr_t)&sX[0][0]);
#endif

    for (int t = 0; t < TT; ++t) {
        const int cb = t & 1, nb = (t + 1) & 1;

#if HAVE_TDM
        if (wv == 0) {
            int tn = (t + 1 < TT) ? (t + 1) : (TT - 1);  // clamped dummy on last iter
            tdm_load_tile(xin + (size_t)(b0 * TT + tn) * II, (unsigned)(uintptr_t)&sX[nb][0]);
            __builtin_amdgcn_s_wait_tensorcnt(1);        // x_t has landed (in-order)
        }
#else
        {   // synchronous fallback: 512 threads x 16B = 8KB tile copy
            uint4 vdat = ((const uint4*)xin)[((size_t)(b0 + (tid >> 5)) * TT + t) * (II / 8) + (tid & 31)];
            ((uint4*)&sX[cb][0])[(tid >> 5) * 32 + (tid & 31)] = vdat;
        }
#endif
        __syncthreads();  // x_t + h_t visible to all waves

        // gates = [x_t | h] (16x512) @ WcatT (512x1024) + bias, this wave's 4 tiles
        f32x8 acc[4];
#pragma unroll
        for (int gi = 0; gi < 4; ++gi) acc[gi] = splat8(bg[gi]);

        // x half: K = 0..255 (rolled: keeps weight loads inside the t-loop)
        {
            const __bf16* xb = &sX[cb][0];
            const __bf16* wk = WT;
#pragma unroll 2
            for (int kk = 0; kk < 8; ++kk) {
                gemm_k32(xb, kk * 32, wk, r, hi, lane, jn, acc);
                wk += 32 * NG;
            }
        }
        // h half: K = 256..511
        {
            const __bf16* hb = &sH[cb][0];
            const __bf16* wk = WT + (size_t)II * NG;
#pragma unroll 2
            for (int kk = 0; kk < 8; ++kk) {
                gemm_k32(hb, kk * 32, wk, r, hi, lane, jn, acc);
                wk += 32 * NG;
            }
        }

        // LSTM cell update directly in accumulator layout: lane col jw, row v+8*hi
#pragma unroll
        for (int v = 0; v < 8; ++v) {
            float gi_ = sigm(acc[0][v]);
            float gf_ = sigm(acc[1][v]);
            float gg_ = tanh_f(acc[2][v]);
            float go_ = sigm(acc[3][v]);
            float c2 = gf_ * c_reg[v] + gi_ * gg_;
            c_reg[v] = c2 * mc[v];                 // masked carried c
            float h2 = go_ * tanh_f(c2);           // uses unmasked c2
            float yv = h2 * mo[v];                 // emitted (out-masked)
            size_t oidx = ((size_t)(b0 + v + hi * 8) * TT + t) * HH + jw;
            if (yf32) yf32[oidx] = yv;
            if (ybf)  ybf[oidx] = (__bf16)yv;
            sH[nb][(v + hi * 8) * 256 + jw] = (__bf16)(h2 * mh[v]);  // carried h
        }
        __syncthreads();  // all reads of sX[cb]/sH[cb] done before next prefetch/write
    }
}

extern "C" void kernel_launch(void* const* d_in, const int* in_sizes, int n_in,
                              void* d_out, int out_size, void* d_ws, size_t ws_size,
                              hipStream_t stream) {
    const float* x     = (const float*)d_in[0];
    const float* Wih   = (const float*)d_in[1];
    const float* Whh   = (const float*)d_in[2];
    const float* bih   = (const float*)d_in[3];
    const float* bhh   = (const float*)d_in[4];
    const float* masks = (const float*)d_in[5];

    // ws layout: [0,64MB) x/y0 bf16 buffer; then packed weights; then bias
    __bf16* xy   = (__bf16*)d_ws;
    __bf16* WT   = (__bf16*)((char*)d_ws + 67108864);            // 2*512*1024 bf16
    float*  bias = (float*)((char*)d_ws + 67108864 + 2097152);   // 2*1024 fp32

    // prep: convert x to bf16, pack/transpose weights + fuse biases
    cvt_x_kernel<<<32768, 256, 0, stream>>>(x, xy);
    pack_w_kernel<<<4096, 256, 0, stream>>>(Wih, Whh, bih, bhh, WT, bias);

    // layer 0: reads x (bf16), writes masked y0 bf16 in place (read-before-write per t)
    lstm_layer_kernel<<<8, 512, 0, stream>>>(xy, xy, nullptr, WT, bias, masks);
    // layer 1: reads y0 bf16, writes final fp32 output
    lstm_layer_kernel<<<8, 512, 0, stream>>>(xy, nullptr, (float*)d_out,
                                             WT + 512 * 1024, bias + NG,
                                             masks + 3 * BB * HH);
}
